// _ProcessorBlock_29669634081498
// MI455X (gfx1250) — matrix operand order repaired and checked by reading, not compiled, not run on hardware
//
#include <hip/hip_runtime.h>
#include <hip/hip_bf16.h>

typedef unsigned short u16;
typedef __attribute__((ext_vector_type(16))) __bf16 v16bf;
typedef __attribute__((ext_vector_type(8)))  float  v8f;

union frag_u { v16bf v; uint4 q[2]; };

// ---------- small helpers ----------
__device__ __forceinline__ u16 f2bf_bits(float f) {
  unsigned u = __float_as_uint(f);
  unsigned r = u + 0x7FFFu + ((u >> 16) & 1u);   // round-to-nearest-even
  return (u16)(r >> 16);
}

__device__ __forceinline__ float gelu_exact(float x) {
  return 0.5f * x * (1.0f + erff(x * 0.70710678118654752440f));
}

// A-fragment (16x32 bf16) from an LDS tile laid out [row][k], k contiguous.
// Per ISA 7.12.2: lane L -> M = L&15 ; lanes 0-15: K = k0+0..7 & k0+16..23,
// lanes 16-31: K = k0+8..15 & k0+24..31.  Two contiguous 16B chunks.
__device__ __forceinline__ v16bf load_a_frag_lds(const u16* tile, int m,
                                                 int stride, int k0, int lane) {
  const int kbase = (lane & 16) ? 8 : 0;
  const u16* p = tile + (size_t)m * stride + k0 + kbase;
  frag_u f;
  f.q[0] = *(const uint4*)(p);
  f.q[1] = *(const uint4*)(p + 16);
  return f.v;
}

// B-fragment (32x16 bf16) from pre-packed weights: layout [frag][lane][16 bf16].
__device__ __forceinline__ v16bf load_b_frag(const u16* wf, int fragIdx, int lane) {
  const uint4* p = (const uint4*)(wf + ((size_t)fragIdx * 32 + lane) * 16);
  frag_u f;
  f.q[0] = p[0];
  f.q[1] = p[1];
  return f.v;
}

// ---------- prep kernels ----------
__global__ void k_f32_to_bf16(const float* __restrict__ x, u16* __restrict__ y, int n) {
  int i = blockIdx.x * blockDim.x + threadIdx.x;
  if (i < n) y[i] = f2bf_bits(x[i]);
}

__global__ void k_zero_f32(float* __restrict__ x, int n) {
  int i = blockIdx.x * blockDim.x + threadIdx.x;
  if (i < n) x[i] = 0.0f;
}

// Pack a KxN (N=128, K multiple of 32) fp32 weight into bf16 B-fragments.
// Fragment f = kt*8 + nt; element (lane,e): k = kt*32 + (e>>3)*16 + ((lane>>4)&1)*8 + (e&7),
// n = nt*16 + (lane&15).  One thread per output bf16.
__global__ void k_pack_wfrag(const float* __restrict__ W, u16* __restrict__ out, int total) {
  int tid = blockIdx.x * blockDim.x + threadIdx.x;
  if (tid >= total) return;
  int f    = tid >> 9;
  int r    = tid & 511;
  int lane = r >> 4;
  int e    = r & 15;
  int kt = f >> 3, nt = f & 7;
  int k = kt * 32 + (e >> 3) * 16 + ((lane >> 4) & 1) * 8 + (e & 7);
  int n = nt * 16 + (lane & 15);
  out[tid] = f2bf_bits(W[(size_t)k * 128 + n]);
}

// ---------- edge kernel: messages + scatter ----------
// 64 edges per workgroup, 256 threads = 8 waves.
// wave (w&3) owns edge rows [(w&3)*16, +16); (w>>2) picks output-col half.
#define ETILE 64
__global__ __launch_bounds__(256)
void k_edge(const u16* __restrict__ hbf, const float* __restrict__ pos,
            const int* __restrict__ srcIdx, const int* __restrict__ dstIdx,
            const u16* __restrict__ We1f, const float* __restrict__ We1last,
            const float* __restrict__ be1, const u16* __restrict__ We2f,
            const float* __restrict__ be2, float* __restrict__ agg) {
  __shared__ __align__(16) u16 Ain[ETILE * 264];    // [hi | hj], K=256, stride 264
  __shared__ __align__(16) u16 Et[ETILE * 136];     // layer-1 output, K=128, stride 136
  __shared__ float dij2[ETILE];
  __shared__ int   dsts[ETILE];

  const int tid = threadIdx.x;
  const int e0  = blockIdx.x * ETILE;

  // ---- gather phase ----
  {
    int m    = tid & 63;
    int part = tid >> 6;                       // 0,1: hi halves; 2,3: hj halves
    int edge = e0 + m;
    int d = dstIdx[edge];
    int s = srcIdx[edge];
    int idx = (part < 2) ? d : s;
    const uint4* sp = (const uint4*)(hbf + (size_t)idx * 128 + (part & 1) * 64);
    uint4* dp = (uint4*)(Ain + (size_t)m * 264 + ((part < 2) ? 0 : 128) + (part & 1) * 64);
#pragma unroll
    for (int i = 0; i < 8; ++i) dp[i] = sp[i];
    if (part == 0) {
      float dx = pos[d * 3 + 0] - pos[s * 3 + 0];
      float dy = pos[d * 3 + 1] - pos[s * 3 + 1];
      float dz = pos[d * 3 + 2] - pos[s * 3 + 2];
      dij2[m] = dx * dx + dy * dy + dz * dz;
      dsts[m] = d;
    }
  }
  __syncthreads();

  const int wave   = tid >> 5;
  const int lane   = tid & 31;
  const int mbase  = (wave & 3) * 16;
  const int ntbase = (wave >> 2) * 4;          // 0 or 4
  const int mA     = mbase + (lane & 15);
  const int ncol   = lane & 15;
  const int moff   = mbase + ((lane & 16) ? 8 : 0);

  // ---- layer 1: e = gelu([hi,hj,dij2] @ We1 + be1) ----
  v8f acc[4];
#pragma unroll
  for (int j = 0; j < 4; ++j) acc[j] = v8f{0.f, 0.f, 0.f, 0.f, 0.f, 0.f, 0.f, 0.f};

#pragma unroll
  for (int kt = 0; kt < 8; ++kt) {
    v16bf a = load_a_frag_lds(Ain, mA, 264, kt * 32, lane);
#pragma unroll
    for (int j = 0; j < 4; ++j) {
      v16bf b = load_b_frag(We1f, kt * 8 + ntbase + j, lane);
      acc[j] = __builtin_amdgcn_wmma_f32_16x16x32_bf16(false, a, false, b,
                                                       (short)0, acc[j], false, false);
    }
  }

#pragma unroll
  for (int j = 0; j < 4; ++j) {
    int   n    = (ntbase + j) * 16 + ncol;
    float w256 = We1last[n];                   // K=257 tail as rank-1 update
    float bias = be1[n];
#pragma unroll
    for (int v = 0; v < 8; ++v) {
      int   m = moff + v;
      float x = acc[j][v] + dij2[m] * w256 + bias;
      Et[(size_t)m * 136 + n] = f2bf_bits(gelu_exact(x));
    }
  }
  __syncthreads();

  // ---- layer 2: e = gelu(e @ We2 + be2), then scatter-add ----
  v8f acc2[4];
#pragma unroll
  for (int j = 0; j < 4; ++j) acc2[j] = v8f{0.f, 0.f, 0.f, 0.f, 0.f, 0.f, 0.f, 0.f};

#pragma unroll
  for (int kt = 0; kt < 4; ++kt) {
    v16bf a = load_a_frag_lds(Et, mA, 136, kt * 32, lane);
#pragma unroll
    for (int j = 0; j < 4; ++j) {
      v16bf b = load_b_frag(We2f, kt * 8 + ntbase + j, lane);
      acc2[j] = __builtin_amdgcn_wmma_f32_16x16x32_bf16(false, a, false, b,
                                                        (short)0, acc2[j], false, false);
    }
  }

#pragma unroll
  for (int j = 0; j < 4; ++j) {
    int   n    = (ntbase + j) * 16 + ncol;
    float bias = be2[n];
#pragma unroll
    for (int v = 0; v < 8; ++v) {
      int   m = moff + v;
      float x = gelu_exact(acc2[j][v] + bias);
      atomicAdd(&agg[(size_t)dsts[m] * 128 + n], x);
    }
  }
}

// ---------- node kernel: update MLP + residual ----------
#define NTILE 64
__global__ __launch_bounds__(256)
void k_node(const u16* __restrict__ hbf, const float* __restrict__ agg,
            const float* __restrict__ h, const u16* __restrict__ Wv1f,
            const float* __restrict__ bv1, const u16* __restrict__ Wv2f,
            const float* __restrict__ bv2, float* __restrict__ out, int Nn) {
  __shared__ __align__(16) u16 Ain[NTILE * 264];   // [h | agg], K=256
  __shared__ __align__(16) u16 Et[NTILE * 136];

  const int tid = threadIdx.x;
  const int r0  = blockIdx.x * NTILE;

  // ---- gather phase ----
  {
    int m    = tid & 63;
    int part = tid >> 6;
    int row  = r0 + m;
    if (row >= Nn) row = Nn - 1;                  // clamp (WMMA needs full EXEC)
    if (part < 2) {
      const uint4* sp = (const uint4*)(hbf + (size_t)row * 128 + part * 64);
      uint4* dp = (uint4*)(Ain + (size_t)m * 264 + part * 64);
#pragma unroll
      for (int i = 0; i < 8; ++i) dp[i] = sp[i];
    } else {
      const float* ap = agg + (size_t)row * 128 + (part & 1) * 64;
      u16* dp = Ain + (size_t)m * 264 + 128 + (part & 1) * 64;
#pragma unroll
      for (int i = 0; i < 64; ++i) dp[i] = f2bf_bits(ap[i]);
    }
  }
  __syncthreads();

  const int wave   = tid >> 5;
  const int lane   = tid & 31;
  const int mbase  = (wave & 3) * 16;
  const int ntbase = (wave >> 2) * 4;
  const int mA     = mbase + (lane & 15);
  const int ncol   = lane & 15;
  const int moff   = mbase + ((lane & 16) ? 8 : 0);

  // ---- layer 1 ----
  v8f acc[4];
#pragma unroll
  for (int j = 0; j < 4; ++j) acc[j] = v8f{0.f, 0.f, 0.f, 0.f, 0.f, 0.f, 0.f, 0.f};

#pragma unroll
  for (int kt = 0; kt < 8; ++kt) {
    v16bf a = load_a_frag_lds(Ain, mA, 264, kt * 32, lane);
#pragma unroll
    for (int j = 0; j < 4; ++j) {
      v16bf b = load_b_frag(Wv1f, kt * 8 + ntbase + j, lane);
      acc[j] = __builtin_amdgcn_wmma_f32_16x16x32_bf16(false, a, false, b,
                                                       (short)0, acc[j], false, false);
    }
  }

#pragma unroll
  for (int j = 0; j < 4; ++j) {
    int   n    = (ntbase + j) * 16 + ncol;
    float bias = bv1[n];
#pragma unroll
    for (int v = 0; v < 8; ++v) {
      int m = moff + v;
      Et[(size_t)m * 136 + n] = f2bf_bits(gelu_exact(acc[j][v] + bias));
    }
  }
  __syncthreads();

  // ---- layer 2 + residual ----
  v8f acc2[4];
#pragma unroll
  for (int j = 0; j < 4; ++j) acc2[j] = v8f{0.f, 0.f, 0.f, 0.f, 0.f, 0.f, 0.f, 0.f};

#pragma unroll
  for (int kt = 0; kt < 4; ++kt) {
    v16bf a = load_a_frag_lds(Et, mA, 136, kt * 32, lane);
#pragma unroll
    for (int j = 0; j < 4; ++j) {
      v16bf b = load_b_frag(Wv2f, kt * 8 + ntbase + j, lane);
      acc2[j] = __builtin_amdgcn_wmma_f32_16x16x32_bf16(false, a, false, b,
                                                        (short)0, acc2[j], false, false);
    }
  }

#pragma unroll
  for (int j = 0; j < 4; ++j) {
    int   n    = (ntbase + j) * 16 + ncol;
    float bias = bv2[n];
#pragma unroll
    for (int v = 0; v < 8; ++v) {
      int m   = moff + v;
      int row = r0 + m;
      if (row < Nn) {
        float x = gelu_exact(acc2[j][v] + bias);
        out[(size_t)row * 128 + n] = h[(size_t)row * 128 + n] + x;
      }
    }
  }
}

// ---------- host launch ----------
extern "C" void kernel_launch(void* const* d_in, const int* in_sizes, int n_in,
                              void* d_out, int out_size, void* d_ws, size_t ws_size,
                              hipStream_t stream) {
  (void)in_sizes; (void)n_in; (void)out_size; (void)ws_size;
  constexpr int N = 20000, H = 128, E = 640000;

  const float* h    = (const float*)d_in[0];
  const float* pos  = (const float*)d_in[1];
  const int*   ei   = (const int*)d_in[2];   // (2,E) int32: row0=src, row1=dst
  const float* We1  = (const float*)d_in[3]; // (257,128)
  const float* be1  = (const float*)d_in[4];
  const float* We2  = (const float*)d_in[5]; // (128,128)
  const float* be2  = (const float*)d_in[6];
  const float* Wv1  = (const float*)d_in[7]; // (256,128)
  const float* bv1  = (const float*)d_in[8];
  const float* Wv2  = (const float*)d_in[9]; // (128,128)
  const float* bv2  = (const float*)d_in[10];
  float* out = (float*)d_out;

  char* ws = (char*)d_ws;
  size_t off = 0;
  auto alloc = [&](size_t bytes) -> void* {
    void* p = ws + off;
    off = (off + bytes + 255) & ~size_t(255);
    return p;
  };
  u16*   hbf  = (u16*)alloc((size_t)N * H * 2);
  float* agg  = (float*)alloc((size_t)N * H * 4);
  u16*   We1f = (u16*)alloc(256 * 128 * 2);
  u16*   We2f = (u16*)alloc(128 * 128 * 2);
  u16*   Wv1f = (u16*)alloc(256 * 128 * 2);
  u16*   Wv2f = (u16*)alloc(128 * 128 * 2);

  int nh = N * H;
  k_f32_to_bf16<<<(nh + 255) / 256, 256, 0, stream>>>(h, hbf, nh);
  k_zero_f32  <<<(nh + 255) / 256, 256, 0, stream>>>(agg, nh);
  k_pack_wfrag<<<(256 * 128) / 256, 256, 0, stream>>>(We1, We1f, 256 * 128);
  k_pack_wfrag<<<(128 * 128) / 256, 256, 0, stream>>>(We2, We2f, 128 * 128);
  k_pack_wfrag<<<(256 * 128) / 256, 256, 0, stream>>>(Wv1, Wv1f, 256 * 128);
  k_pack_wfrag<<<(128 * 128) / 256, 256, 0, stream>>>(Wv2, Wv2f, 128 * 128);

  k_edge<<<E / ETILE, 256, 0, stream>>>(hbf, pos, ei, ei + E, We1f,
                                        We1 + 256 * 128, be1, We2f, be2, agg);
  k_node<<<(N + NTILE - 1) / NTILE, 256, 0, stream>>>(hbf, agg, h, Wv1f, bv1,
                                                      Wv2f, bv2, out, N);
}
